// Attention_71751723647784
// MI455X (gfx1250) — compile-verified, matrix operand order checked by reading, not run on hardware
//
#include <hip/hip_runtime.h>
#include <hip/hip_bf16.h>
#include <math.h>

typedef __attribute__((ext_vector_type(16))) _Float16 v16h;
typedef __attribute__((ext_vector_type(8)))  _Float16 v8h;
typedef __attribute__((ext_vector_type(8)))  float    v8f;

#define Qn   2048
#define Kn   2048
#define Cin  128
#define Hh   8
#define CHd  32
#define HDd  256

// ---- WMMA 16x32 f16 operand loader (ISA 7.12.2 16-bit A layout) ----------
// lane&15 = row; kb=(lane>>4)*8; halves 0..7 -> k=kb+0..7, 8..15 -> k=kb+16..23
// Two contiguous 16B vector loads -> no per-half packing moves.
static __device__ inline v16h load_tile16(const _Float16* p, int stride, int lane) {
  const int row = lane & 15;
  const int kb  = (lane >> 4) << 3;
  const _Float16* r = p + (long)row * stride + kb;
  v8h lo = *(const v8h*)(r);
  v8h hi = *(const v8h*)(r + 16);
  return __builtin_shufflevector(lo, hi, 0, 1, 2, 3, 4, 5, 6, 7,
                                         8, 9, 10, 11, 12, 13, 14, 15);
}

static __device__ inline float rmax16(float v) {
  v = fmaxf(v, __shfl_xor(v, 1, 32));
  v = fmaxf(v, __shfl_xor(v, 2, 32));
  v = fmaxf(v, __shfl_xor(v, 4, 32));
  v = fmaxf(v, __shfl_xor(v, 8, 32));
  return v;
}
static __device__ inline float radd16(float v) {
  v += __shfl_xor(v, 1, 32);
  v += __shfl_xor(v, 2, 32);
  v += __shfl_xor(v, 4, 32);
  v += __shfl_xor(v, 8, 32);
  return v;
}

// ---------------------------- conversions ----------------------------------
__global__ void cvt_f32_f16(const float* __restrict__ s, _Float16* __restrict__ d,
                            int n, float scale) {
  int i = blockIdx.x * blockDim.x + threadIdx.x;
  if (i < n) d[i] = (_Float16)(s[i] * scale);
}

// [R,C] f32 -> transposed [C,R] f16 (weights are tiny; done once)
__global__ void cvt_f32_f16_t(const float* __restrict__ s, _Float16* __restrict__ d,
                              int R, int C, float scale) {
  int i = blockIdx.x * blockDim.x + threadIdx.x;
  if (i < R * C) {
    int r = i / C, c = i % C;
    d[(long)c * R + r] = (_Float16)(s[i] * scale);
  }
}

// ------------- GEMM: Y = X[M,Kd] @ Wt[N,Kd]^T  (all operands f16) ----------
// 256 thr = 8 waves; block tile 64x64; wave 16x32 (two 16x16 WMMA accs)
// mode 0: row-major f16   mode 1: transposed f16 (Y[col*M+row], for V)
// mode 2: sigmoid(x + bvec[col]) row-major f16 (gate)
__global__ void proj_gemm(const _Float16* __restrict__ X, const _Float16* __restrict__ Wt,
                          const float* __restrict__ bvec, _Float16* __restrict__ Y,
                          int M, int N, int Kd, int mode) {
  const int lane = threadIdx.x & 31;
  const int w    = threadIdx.x >> 5;
  const int m0 = blockIdx.x * 64 + (w >> 1) * 16;
  const int n0 = blockIdx.y * 64 + (w & 1) * 32;
  v8f acc0 = {}; v8f acc1 = {};
  for (int kk = 0; kk < Kd; kk += 32) {
    v16h a  = load_tile16(X  + (long)m0 * Kd + kk, Kd, lane);
    v16h b0 = load_tile16(Wt + (long)n0 * Kd + kk, Kd, lane);
    v16h b1 = load_tile16(Wt + (long)(n0 + 16) * Kd + kk, Kd, lane);
    acc0 = __builtin_amdgcn_wmma_f32_16x16x32_f16(false, a, false, b0, (short)0, acc0, false, false);
    acc1 = __builtin_amdgcn_wmma_f32_16x16x32_f16(false, a, false, b1, (short)0, acc1, false, false);
  }
  const int g = lane >> 4, n = lane & 15;
#pragma unroll
  for (int j = 0; j < 8; ++j) {
    int row = m0 + 8 * g + j;
    int c0 = n0 + n, c1 = n0 + 16 + n;
    float v0 = acc0[j], v1 = acc1[j];
    if (mode == 2) {
      v0 = __builtin_amdgcn_rcpf(1.0f + __expf(-(v0 + bvec[c0])));
      v1 = __builtin_amdgcn_rcpf(1.0f + __expf(-(v1 + bvec[c1])));
    }
    if (mode == 1) {
      Y[(long)c0 * M + row] = (_Float16)v0;
      Y[(long)c1 * M + row] = (_Float16)v1;
    } else {
      Y[(long)row * N + c0] = (_Float16)v0;
      Y[(long)row * N + c1] = (_Float16)v1;
    }
  }
}

// --------------------- fused flash attention -------------------------------
// grid.x = Qn/16 ; 256 thr = 8 waves; wave h = head h, 16-row q tile,
// 64-key inner tile (8 WMMAs/iter).  Streams bias+distance exactly once.
__global__ void attn_fused(const _Float16* __restrict__ Qp, const _Float16* __restrict__ Kp,
                           const _Float16* __restrict__ Vt, const float* __restrict__ bias,
                           const float* __restrict__ dist, const _Float16* __restrict__ Gp,
                           _Float16* __restrict__ Otmp) {
  __shared__ _Float16 plds[Hh * 16 * 64];   // per-wave 2KB P-transpose slab
  const int lane = threadIdx.x & 31;
  const int h    = threadIdx.x >> 5;
  const int q0   = blockIdx.x * 16;
  const int g = lane >> 4, n = lane & 15;
  _Float16* myp = plds + h * (16 * 64);

  const v16h aq = load_tile16(Qp + (long)q0 * HDd + h * CHd, HDd, lane);

  float mrow[8], lrow[8];
#pragma unroll
  for (int j = 0; j < 8; ++j) { mrow[j] = -1e30f; lrow[j] = 0.0f; }
  v8f o0 = {}; v8f o1 = {};

  for (int k0 = 0; k0 < Kn; k0 += 64) {
    // S = Q*K^T : four 16x16 C tiles over 64 keys (contraction = CH = 32)
    v8f s[4];
#pragma unroll
    for (int t = 0; t < 4; ++t) {
      v16h bk = load_tile16(Kp + (long)(k0 + 16 * t) * HDd + h * CHd, HDd, lane);
      v8f z = {};
      s[t] = __builtin_amdgcn_wmma_f32_16x16x32_f16(false, aq, false, bk, (short)0, z, false, false);
    }

    // prefetch next tile of the dominant streams (global_prefetch_b8)
    if (k0 + 64 < Kn) {
      const long nb = (long)(q0 + 8 * g) * Kn + (k0 + 64);
      __builtin_prefetch(dist + (nb + n * 4) * Hh + h, 0, 1);
      __builtin_prefetch(bias + nb + n * 4, 0, 1);
    }

    // bias (L2-amplified across heads) + distance (the HBM stream)
#pragma unroll
    for (int t = 0; t < 4; ++t) {
#pragma unroll
      for (int j = 0; j < 8; ++j) {
        const long base = (long)(q0 + 8 * g + j) * Kn + k0 + 16 * t + n;
        s[t][j] += bias[base] + dist[base * Hh + h];
      }
    }

    // online softmax over the 64-key tile
#pragma unroll
    for (int j = 0; j < 8; ++j) {
      float tm = fmaxf(fmaxf(s[0][j], s[1][j]), fmaxf(s[2][j], s[3][j]));
      tm = rmax16(tm);
      float mn = fmaxf(mrow[j], tm);
      float sc = __expf(mrow[j] - mn);
      float p0 = __expf(s[0][j] - mn);
      float p1 = __expf(s[1][j] - mn);
      float p2 = __expf(s[2][j] - mn);
      float p3 = __expf(s[3][j] - mn);
      lrow[j] = lrow[j] * sc + radd16((p0 + p1) + (p2 + p3));
      mrow[j] = mn;
      o0[j] *= sc; o1[j] *= sc;
      const int r = (8 * g + j) * 64 + n;
      myp[r]      = (_Float16)p0;
      myp[r + 16] = (_Float16)p1;
      myp[r + 32] = (_Float16)p2;
      myp[r + 48] = (_Float16)p3;
    }

    // P back in A layout (per-wave LDS ops are in-order; private slab)
    v16h ap0 = load_tile16(myp,      64, lane);   // keys k0 .. k0+31
    v16h ap1 = load_tile16(myp + 32, 64, lane);   // keys k0+32 .. k0+63
    // V transposed [H*CH][Kn]: contiguous 16B loads along keys
    const _Float16* vb = Vt + (long)(h * CHd) * Kn + k0;
    v16h bv00 = load_tile16(vb,                Kn, lane);  // ch 0-15 , keys +0
    v16h bv01 = load_tile16(vb + 32,           Kn, lane);  // ch 0-15 , keys +32
    v16h bv10 = load_tile16(vb + 16 * Kn,      Kn, lane);  // ch 16-31, keys +0
    v16h bv11 = load_tile16(vb + 16 * Kn + 32, Kn, lane);  // ch 16-31, keys +32
    o0 = __builtin_amdgcn_wmma_f32_16x16x32_f16(false, ap0, false, bv00, (short)0, o0, false, false);
    o0 = __builtin_amdgcn_wmma_f32_16x16x32_f16(false, ap1, false, bv01, (short)0, o0, false, false);
    o1 = __builtin_amdgcn_wmma_f32_16x16x32_f16(false, ap0, false, bv10, (short)0, o1, false, false);
    o1 = __builtin_amdgcn_wmma_f32_16x16x32_f16(false, ap1, false, bv11, (short)0, o1, false, false);
  }

  // normalize, gate, store f16 for the output GEMM
#pragma unroll
  for (int j = 0; j < 8; ++j) {
    const int q = q0 + 8 * g + j;
    const float inv = __builtin_amdgcn_rcpf(lrow[j]);
    const long base = (long)q * HDd + h * CHd;
    float g0 = (float)Gp[base + n];
    float g1 = (float)Gp[base + 16 + n];
    Otmp[base + n]      = (_Float16)(o0[j] * inv * g0);
    Otmp[base + 16 + n] = (_Float16)(o1[j] * inv * g1);
  }
}

// ---------------- output GEMM: out = O @ Wo + bo (f32 out) -----------------
__global__ void out_gemm(const _Float16* __restrict__ X, const _Float16* __restrict__ Wt,
                         const float* __restrict__ bo, float* __restrict__ out,
                         int M, int N, int Kd) {
  const int lane = threadIdx.x & 31;
  const int w    = threadIdx.x >> 5;
  const int m0 = blockIdx.x * 64 + (w >> 1) * 16;
  const int n0 = blockIdx.y * 64 + (w & 1) * 32;
  v8f acc0 = {}; v8f acc1 = {};
  for (int kk = 0; kk < Kd; kk += 32) {
    v16h a  = load_tile16(X  + (long)m0 * Kd + kk, Kd, lane);
    v16h b0 = load_tile16(Wt + (long)n0 * Kd + kk, Kd, lane);
    v16h b1 = load_tile16(Wt + (long)(n0 + 16) * Kd + kk, Kd, lane);
    acc0 = __builtin_amdgcn_wmma_f32_16x16x32_f16(false, a, false, b0, (short)0, acc0, false, false);
    acc1 = __builtin_amdgcn_wmma_f32_16x16x32_f16(false, a, false, b1, (short)0, acc1, false, false);
  }
  const int g = lane >> 4, n = lane & 15;
#pragma unroll
  for (int j = 0; j < 8; ++j) {
    int row = m0 + 8 * g + j;
    int c0 = n0 + n, c1 = n0 + 16 + n;
    out[(long)row * N + c0] = acc0[j] + bo[c0];
    out[(long)row * N + c1] = acc1[j] + bo[c1];
  }
}

// ---------------------------------------------------------------------------
extern "C" void kernel_launch(void* const* d_in, const int* in_sizes, int n_in,
                              void* d_out, int out_size, void* d_ws, size_t ws_size,
                              hipStream_t stream) {
  (void)in_sizes; (void)n_in; (void)out_size; (void)ws_size;
  const float* q_x  = (const float*)d_in[0];
  const float* kv_x = (const float*)d_in[1];
  const float* bias = (const float*)d_in[2];
  const float* dist = (const float*)d_in[3];
  const float* Wq   = (const float*)d_in[4];
  const float* Wk   = (const float*)d_in[5];
  const float* Wv   = (const float*)d_in[6];
  const float* Wg   = (const float*)d_in[7];
  const float* bg   = (const float*)d_in[8];
  const float* Wo   = (const float*)d_in[9];
  const float* bo   = (const float*)d_in[10];

  char* ws = (char*)d_ws;
  _Float16* qx16  = (_Float16*)(ws + 0);            // 2048*128
  _Float16* kv16  = (_Float16*)(ws + 524288);       // 2048*128
  _Float16* wq16t = (_Float16*)(ws + 1048576);      // [256][128]
  _Float16* wk16t = (_Float16*)(ws + 1114112);
  _Float16* wv16t = (_Float16*)(ws + 1179648);
  _Float16* wg16t = (_Float16*)(ws + 1245184);
  _Float16* wo16t = (_Float16*)(ws + 1310720);      // [128][256]
  _Float16* Qp    = (_Float16*)(ws + 1376256);      // [2048][256]
  _Float16* Kp    = (_Float16*)(ws + 2424832);      // [2048][256]
  _Float16* Vt    = (_Float16*)(ws + 3473408);      // [256][2048] (transposed)
  _Float16* Gp    = (_Float16*)(ws + 4521984);      // [2048][256]
  _Float16* Ot    = (_Float16*)(ws + 5570560);      // [2048][256]

  const int nAct = Qn * Cin;        // 262144
  const int nW   = Cin * HDd;       // 32768
  const float qscale = 0.17677669529663687f;  // 1/sqrt(32), folded into Wq

  cvt_f32_f16<<<(nAct + 255) / 256, 256, 0, stream>>>(q_x,  qx16, nAct, 1.0f);
  cvt_f32_f16<<<(nAct + 255) / 256, 256, 0, stream>>>(kv_x, kv16, nAct, 1.0f);
  cvt_f32_f16_t<<<(nW + 255) / 256, 256, 0, stream>>>(Wq, wq16t, Cin, HDd, qscale);
  cvt_f32_f16_t<<<(nW + 255) / 256, 256, 0, stream>>>(Wk, wk16t, Cin, HDd, 1.0f);
  cvt_f32_f16_t<<<(nW + 255) / 256, 256, 0, stream>>>(Wv, wv16t, Cin, HDd, 1.0f);
  cvt_f32_f16_t<<<(nW + 255) / 256, 256, 0, stream>>>(Wg, wg16t, Cin, HDd, 1.0f);
  cvt_f32_f16_t<<<(nW + 255) / 256, 256, 0, stream>>>(Wo, wo16t, HDd, Cin, 1.0f);

  dim3 gp(Qn / 64, HDd / 64);   // 32 x 4
  proj_gemm<<<gp, 256, 0, stream>>>(qx16, wq16t, nullptr, Qp, Qn, HDd, Cin, 0);
  proj_gemm<<<gp, 256, 0, stream>>>(kv16, wk16t, nullptr, Kp, Kn, HDd, Cin, 0);
  proj_gemm<<<gp, 256, 0, stream>>>(kv16, wv16t, nullptr, Vt, Kn, HDd, Cin, 1);
  proj_gemm<<<gp, 256, 0, stream>>>(qx16, wg16t, bg,      Gp, Qn, HDd, Cin, 2);

  attn_fused<<<Qn / 16, 256, 0, stream>>>(Qp, Kp, Vt, bias, dist, Gp, Ot);

  dim3 go(Qn / 64, Cin / 64);   // 32 x 2
  out_gemm<<<go, 256, 0, stream>>>(Ot, wo16t, bo, (float*)d_out, Qn, Cin, HDd);
}
